// MultiHeadAttentionLayer_1563368096363
// MI455X (gfx1250) — matrix operand order repaired and checked by reading, not compiled, not run on hardware
//
#include <hip/hip_runtime.h>
#include <hip/hip_bf16.h>

#define D_MODEL 1024
#define SEQ     2048
#define BATCH   2
#define HEADS   16
#define HDIM    64

typedef __attribute__((ext_vector_type(16))) __bf16 v16bf;
typedef __attribute__((ext_vector_type(8)))  __bf16 v8bf;
typedef __attribute__((ext_vector_type(8)))  float  v8f;

static __device__ __forceinline__ v8f wmma_bf16(v16bf a, v16bf b, v8f c) {
    return __builtin_amdgcn_wmma_f32_16x16x32_bf16(
        /*neg_a=*/false, a, /*neg_b=*/false, b,
        /*c_mod=*/(short)0, c, /*reuse_a=*/false, /*reuse_b=*/false);
}

// A-fragment (16x32 bf16): lane holds row = lane&15,
// elements 0..15 = K (lane>>4)*16 + 0..15
static __device__ __forceinline__ v16bf load_a_frag(const __bf16* rowptr, int hi8) {
    v8bf lo = *(const v8bf*)(rowptr + hi8 * 2);
    v8bf hv = *(const v8bf*)(rowptr + hi8 * 2 + 8);
    v16bf r;
#pragma unroll
    for (int i = 0; i < 8; ++i) { r[i] = lo[i]; r[i + 8] = hv[i]; }
    return r;
}

static __device__ __forceinline__ v16bf combine16(v8bf lo, v8bf hv) {
    v16bf r;
#pragma unroll
    for (int i = 0; i < 8; ++i) { r[i] = lo[i]; r[i + 8] = hv[i]; }
    return r;
}

static __device__ __forceinline__ v8f zero8() {
    v8f z;
#pragma unroll
    for (int i = 0; i < 8; ++i) z[i] = 0.0f;
    return z;
}

// ---------------- Kernel 0: f32 -> bf16 conversion of x and W ----------------
__global__ __launch_bounds__(256) void cvt_kernel(const float* __restrict__ x,
                                                  const float* __restrict__ W,
                                                  __bf16* __restrict__ xb,
                                                  __bf16* __restrict__ Wb,
                                                  int nx, int nw) {
    int i = blockIdx.x * blockDim.x + threadIdx.x;
    if (i < nx) xb[i] = (__bf16)x[i];
    if (i < nw) Wb[i] = (__bf16)W[i];
}

// ---------------- Kernel 1: QKV = x @ W^T + b, scattered to q/k/vT ----------------
// Block: 256 thr (8 waves). Block tile: 128 rows x 64 cols.
// W tile (64 x 32 bf16) staged in LDS, double buffered with STATIC buffer indices
// (manual 2x unroll, branch-free clamped prefetch -> no phi register shuffling).
#define BPAD 40   // 32 K elements + 8 pad (bank spread)
__global__ __launch_bounds__(256) void qkv_kernel(const __bf16* __restrict__ xb,
                                                  const __bf16* __restrict__ Wb,
                                                  const float*  __restrict__ bias,
                                                  __bf16* __restrict__ qws,
                                                  __bf16* __restrict__ kws,
                                                  __bf16* __restrict__ vtws) {
    __shared__ __align__(16) __bf16 Bt[2][64][BPAD];

    const int tid   = threadIdx.x;
    const int lane  = tid & 31;
    const int w     = tid >> 5;
    const int lo16  = lane & 15;
    const int hi    = lane >> 4;
    const int rbase = blockIdx.x * 128 + w * 16;
    const int nbase = blockIdx.y * 64;

    // cooperative W-tile loader: thread -> (row n, 8-elem chunk of K)
    const int ln = tid >> 2;          // 0..63
    const int lk = (tid & 3) * 8;     // 0,8,16,24
    const __bf16* wrow = Wb + (size_t)(nbase + ln) * D_MODEL + lk;

    v8f acc[4];
#pragma unroll
    for (int ni = 0; ni < 4; ++ni) acc[ni] = zero8();

    const __bf16* arow = xb + (size_t)(rbase + lo16) * D_MODEL;
    const int hi8 = hi * 8;

    // prologue: stage tiles 0 and 1
    {
        v8bf b0 = *(const v8bf*)(wrow);
        v8bf b1 = *(const v8bf*)(wrow + 32);
        *(v8bf*)&Bt[0][ln][lk] = b0;
        *(v8bf*)&Bt[1][ln][lk] = b1;
    }
    v16bf a0 = load_a_frag(arow, hi8);
    v16bf a1 = load_a_frag(arow + 32, hi8);
    __syncthreads();

    const __bf16* bfp = &Bt[0][lo16][hi * 16];   // wave-local B-frag base (buf0)
    const int bstride = 16 * BPAD;               // elements between ni tiles
    const int bufstride = 64 * BPAD;             // elements between buffers

#pragma unroll 1
    for (int kt = 0; kt < D_MODEL / 32; kt += 2) {
        // ---------- half A: consume buf0 (tile kt), prefetch tile kt+2 ----------
        const int kA = (kt + 2 <= 30) ? (kt + 2) : 30;   // clamped: last iter reloads dead tile
        v8bf  bnA = *(const v8bf*)(wrow + kA * 32);
        v16bf anA = load_a_frag(arow + kA * 32, hi8);
#pragma unroll
        for (int ni = 0; ni < 4; ++ni) {
            const __bf16* bp = bfp + ni * bstride;
            v16bf bf = combine16(*(const v8bf*)bp, *(const v8bf*)(bp + 8));
            acc[ni] = wmma_bf16(a0, bf, acc[ni]);
        }
        __syncthreads();                          // all waves done reading buf0
        *(v8bf*)&Bt[0][ln][lk] = bnA;             // refill buf0 for kt+2
        a0 = anA;

        // ---------- half B: consume buf1 (tile kt+1), prefetch tile kt+3 ----------
        const int kB = (kt + 3 <= 31) ? (kt + 3) : 31;
        v8bf  bnB = *(const v8bf*)(wrow + kB * 32);
        v16bf anB = load_a_frag(arow + kB * 32, hi8);
#pragma unroll
        for (int ni = 0; ni < 4; ++ni) {
            const __bf16* bp = bfp + bufstride + ni * bstride;
            v16bf bf = combine16(*(const v8bf*)bp, *(const v8bf*)(bp + 8));
            acc[ni] = wmma_bf16(a1, bf, acc[ni]);
        }
        __syncthreads();                          // all waves done reading buf1
        *(v8bf*)&Bt[1][ln][lk] = bnB;             // refill buf1 for kt+3
        a1 = anB;
    }
    __syncthreads();                              // protect trailing stores (paranoia)

    // Epilogue: add bias, scatter bf16 into q/k/vT layouts
#pragma unroll
    for (int ni = 0; ni < 4; ++ni) {
        const int n     = nbase + ni * 16 + lo16;   // col in [0, 3*D)
        const float bv  = bias[n];
        const int which = n >> 10;                  // 0=q 1=k 2=v
        const int d     = n & (D_MODEL - 1);
        const int h     = d >> 6;
        const int hd_i  = d & (HDIM - 1);
#pragma unroll
        for (int r = 0; r < 8; ++r) {
            const int s_g = rbase + r + 8 * hi;     // global row in [0, B*S)
            const int b_i = s_g >> 11;
            const int s_l = s_g & (SEQ - 1);
            const int bh  = b_i * HEADS + h;
            const __bf16 v = (__bf16)(acc[ni][r] + bv);
            if (which == 0)
                qws[((size_t)bh * SEQ + s_l) * HDIM + hd_i] = v;
            else if (which == 1)
                kws[((size_t)bh * SEQ + s_l) * HDIM + hd_i] = v;
            else
                vtws[((size_t)bh * HDIM + hd_i) * SEQ + s_l] = v;
        }
    }
}

// ---------------- Kernel 2: causal flash attention + residual ----------------
// grid = (SEQ/128, B*H); block 256 (8 waves); wave owns 16 queries.
__global__ __launch_bounds__(256) void attn_kernel(const __bf16* __restrict__ qws,
                                                   const __bf16* __restrict__ kws,
                                                   const __bf16* __restrict__ vtws,
                                                   const float*  __restrict__ x,
                                                   float* __restrict__ out) {
    __shared__ __align__(16) __bf16 Plds[8][16 * 64];

    const int lane  = threadIdx.x & 31;
    const int w     = threadIdx.x >> 5;
    const int lo16  = lane & 15;
    const int hi    = lane >> 4;
    const int hi8   = hi * 8;
    const int bh    = blockIdx.y;
    const int qbase = blockIdx.x * 128 + w * 16;    // first query row of this wave

    const __bf16* qp = qws  + (size_t)bh * SEQ * HDIM;
    const __bf16* kp = kws  + (size_t)bh * SEQ * HDIM;
    const __bf16* vp = vtws + (size_t)bh * HDIM * SEQ;

    float m[8], l[8];
    v8f o[4];
#pragma unroll
    for (int r = 0; r < 8; ++r) { m[r] = -__builtin_inff(); l[r] = 0.0f; }
#pragma unroll
    for (int ni = 0; ni < 4; ++ni) o[ni] = zero8();

    // Q A-fragments are loop-invariant: load once.
    const __bf16* qrow = qp + (size_t)(qbase + lo16) * HDIM;
    v16bf aq[2];
#pragma unroll
    for (int kk = 0; kk < 2; ++kk) aq[kk] = load_a_frag(qrow + kk * 32, hi8);

    const int jlast = qbase >> 6;                   // causal: key tile <= query tile

    for (int j = 0; j <= jlast; ++j) {
        // ---- scores S = Q K^T  (16 x 64, K = 64) ----
        v8f sc[4];
#pragma unroll
        for (int ni = 0; ni < 4; ++ni) sc[ni] = zero8();
#pragma unroll
        for (int kk = 0; kk < 2; ++kk) {
#pragma unroll
            for (int ni = 0; ni < 4; ++ni) {
                const int key = j * 64 + ni * 16 + lo16;
                v16bf bf = *(const v16bf*)(kp + (size_t)key * HDIM + kk * 32 + hi * 16);
                sc[ni] = wmma_bf16(aq[kk], bf, sc[ni]);
            }
        }
        // ---- issue V fragment loads now; softmax VALU work hides the latency ----
        v16bf vfr[2][4];
#pragma unroll
        for (int kk = 0; kk < 2; ++kk)
#pragma unroll
            for (int ni = 0; ni < 4; ++ni) {
                const int n = ni * 16 + lo16;
                vfr[kk][ni] = *(const v16bf*)(vp + (size_t)n * SEQ + j * 64 + kk * 32 + hi * 16);
            }
        // ---- causal mask on the diagonal tile ----
        if (j == jlast) {
#pragma unroll
            for (int ni = 0; ni < 4; ++ni) {
                const int col = j * 64 + ni * 16 + lo16;
#pragma unroll
                for (int r = 0; r < 8; ++r) {
                    const int rowg = qbase + r + 8 * hi;
                    if (col > rowg) sc[ni][r] = -__builtin_inff();
                }
            }
        }
        // ---- online softmax update (row stats across 16 lanes of each half-wave) ----
#pragma unroll
        for (int r = 0; r < 8; ++r) {
            float t = fmaxf(fmaxf(sc[0][r], sc[1][r]), fmaxf(sc[2][r], sc[3][r]));
#pragma unroll
            for (int msk = 1; msk < 16; msk <<= 1)
                t = fmaxf(t, __shfl_xor(t, msk, 32));
            const float mnew  = fmaxf(m[r], t);
            const float scale = __expf(m[r] - mnew);
            float psum = 0.0f;
#pragma unroll
            for (int ni = 0; ni < 4; ++ni) {
                const float p = __expf(sc[ni][r] - mnew);
                sc[ni][r] = p;
                psum += p;
            }
#pragma unroll
            for (int msk = 1; msk < 16; msk <<= 1)
                psum += __shfl_xor(psum, msk, 32);
            l[r] = l[r] * scale + psum;
            m[r] = mnew;
#pragma unroll
            for (int ni = 0; ni < 4; ++ni) o[ni][r] *= scale;
        }
        // ---- P (C-layout) -> LDS -> A-layout bf16 fragments ----
        __bf16* pl = &Plds[w][0];
#pragma unroll
        for (int ni = 0; ni < 4; ++ni)
#pragma unroll
            for (int r = 0; r < 8; ++r)
                pl[(r + 8 * hi) * 64 + ni * 16 + lo16] = (__bf16)sc[ni][r];
        asm volatile("s_wait_dscnt 0" ::: "memory");   // same-wave LDS RAW
        // ---- O += P V  (K = 64 keys) ----
#pragma unroll
        for (int kk = 0; kk < 2; ++kk) {
            v16bf a = load_a_frag(pl + lo16 * 64 + kk * 32, hi8);
#pragma unroll
            for (int ni = 0; ni < 4; ++ni)
                o[ni] = wmma_bf16(a, vfr[kk][ni], o[ni]);
        }
    }

    // ---- normalize, residual, store f32 ----
    const int b_i = bh >> 4;
    const int h   = bh & (HEADS - 1);
#pragma unroll
    for (int r = 0; r < 8; ++r) {
        const float inv = 1.0f / l[r];
        const int s_g = qbase + r + 8 * hi;
        const size_t rowoff = ((size_t)b_i * SEQ + s_g) * D_MODEL + h * HDIM;
#pragma unroll
        for (int ni = 0; ni < 4; ++ni) {
            const int n = ni * 16 + lo16;
            out[rowoff + n] = x[rowoff + n] + o[ni][r] * inv;
        }
    }
}

extern "C" void kernel_launch(void* const* d_in, const int* in_sizes, int n_in,
                              void* d_out, int out_size, void* d_ws, size_t ws_size,
                              hipStream_t stream) {
    const float* x    = (const float*)d_in[0];   // [2,2048,1024]
    const float* W    = (const float*)d_in[1];   // [3072,1024]
    const float* bias = (const float*)d_in[2];   // [3072]
    float* out = (float*)d_out;

    const int NX = BATCH * SEQ * D_MODEL;        // 4,194,304
    const int NW = 3 * D_MODEL * D_MODEL;        // 3,145,728

    __bf16* xb  = (__bf16*)d_ws;                 // NX bf16
    __bf16* Wb  = xb + NX;                       // NW bf16
    __bf16* qws = Wb + NW;                       // NX bf16 [B,H,S,64]
    __bf16* kws = qws + NX;                      // NX bf16 [B,H,S,64]
    __bf16* vt  = kws + NX;                      // NX bf16 [B,H,64,S]

    cvt_kernel<<<(NX + 255) / 256, 256, 0, stream>>>(x, W, xb, Wb, NX, NW);

    dim3 g1((BATCH * SEQ) / 128, (3 * D_MODEL) / 64);   // (32, 48)
    qkv_kernel<<<g1, 256, 0, stream>>>(xb, Wb, bias, qws, kws, vt);

    dim3 g2(SEQ / 128, BATCH * HEADS);                  // (16, 32)
    attn_kernel<<<g2, 256, 0, stream>>>(qws, kws, vt, x, out);
}
